// MultiHeadAttention_84353157693730
// MI455X (gfx1250) — compile-verified, matrix operand order checked by reading, not run on hardware
//
#include <hip/hip_runtime.h>
#include <math.h>

// ---------------- problem constants ----------------
#define BB    4
#define TSEQ  2048
#define DMODEL 1024
#define NH    16
#define HD    64
#define GM    (BB*TSEQ)   // 8192 rows for projections
#define GK    DMODEL
#define GN    DMODEL
#define SCALE_F 0.125f    // 64^-0.5

// ---------------- WMMA types / helpers ----------------
typedef __attribute__((ext_vector_type(4)))  _Float16 v4h;
typedef __attribute__((ext_vector_type(8)))  _Float16 v8h;
typedef __attribute__((ext_vector_type(16))) _Float16 v16h;
typedef __attribute__((ext_vector_type(8)))  float    v8f;

__device__ __forceinline__ v8f wmma32f16(v16h a, v16h b, v8f c) {
    // D = A(16x32 f16) * B(32x16 f16) + C(16x16 f32)
    return __builtin_amdgcn_wmma_f32_16x16x32_f16(
        /*neg_a=*/false, a, /*neg_b=*/false, b,
        /*c_mod=*/(short)0, c, /*reuse_a=*/false, /*reuse_b=*/false);
}

// Load one A- or B-fragment (16x32 / 32x16 f16) for this lane.
// ISA layout: elements 0..7  <- k = 8*g + 0..7
//             elements 8..15 <- k = 16 + 8*g + 0..7   (g = lane>>4)
// p must point at (row/col base) + k-window start; 16B aligned.
__device__ __forceinline__ v16h load_frag(const _Float16* p, int g) {
    union { v16h v; v8h h[2]; } u;
    u.h[0] = *(const v8h*)(p + 8*g);
    u.h[1] = *(const v8h*)(p + 16 + 8*g);
    return u.v;
}

// ---------------- GEMM:  C = A @ W^T + bias ----------------
// A: M x K (f32 or f16), W: N x K f32 (torch Linear weight), row-major.
// SPLIT: scatter output to (B,H,T,Hd) f16; else dense f32 (M x N).
// 256 threads = 8 waves; tile 128x128, K-step 32; wave -> 16x128 strip
// (8 C-fragments, single A-fragment reused 8x). Double-buffered LDS.
#define BM 128
#define BN 128
#define BK 32
#define NF 8     // N-fragments per wave
#define LDA 40   // padded LDS stride (halves); 80B rows keep 16B alignment
#define NKT (GK / BK)

template<bool A_HALF, bool SPLIT>
__global__ __launch_bounds__(256)
void wmma_gemm_bias(const void* __restrict__ Ap, const float* __restrict__ W,
                    const float* __restrict__ bias, void* __restrict__ Cp)
{
    __shared__ _Float16 As[2][BM * LDA];
    __shared__ _Float16 Bs[2][BN * LDA];

    const int tid  = threadIdx.x;
    const int w    = tid >> 5;
    const int lane = tid & 31;
    const int g    = lane >> 4;
    const int ln   = lane & 15;
    const int m0   = blockIdx.y * BM;
    const int n0   = blockIdx.x * BN;

    const v8f zf = {0.f,0.f,0.f,0.f,0.f,0.f,0.f,0.f};
    v8f acc[NF];
    #pragma unroll
    for (int f = 0; f < NF; ++f) acc[f] = zf;

    const float*    Af = (const float*)Ap;
    const _Float16* Ah = (const _Float16*)Ap;

    // stage tile kt into LDS buffer buf
    auto stage = [&](int kt, int buf) {
        const int k0 = kt * BK;
        if (A_HALF) {
            #pragma unroll
            for (int i = tid; i < (BM*BK)/8; i += 256) {
                const int e = i*8, r = e / BK, c = e % BK;
                *(v8h*)(&As[buf][r*LDA + c]) =
                    *(const v8h*)(Ah + (size_t)(m0 + r)*GK + k0 + c);
            }
        } else {
            #pragma unroll
            for (int i = tid; i < (BM*BK)/4; i += 256) {
                const int e = i*4, r = e / BK, c = e % BK;
                const float4 x = *(const float4*)(Af + (size_t)(m0 + r)*GK + k0 + c);
                const v4h y = { (_Float16)x.x, (_Float16)x.y,
                                (_Float16)x.z, (_Float16)x.w };
                *(v4h*)(&As[buf][r*LDA + c]) = y;
            }
        }
        #pragma unroll
        for (int i = tid; i < (BN*BK)/4; i += 256) {
            const int e = i*4, r = e / BK, c = e % BK;
            const float4 x = *(const float4*)(W + (size_t)(n0 + r)*GK + k0 + c);
            const v4h y = { (_Float16)x.x, (_Float16)x.y,
                            (_Float16)x.z, (_Float16)x.w };
            *(v4h*)(&Bs[buf][r*LDA + c]) = y;
        }
    };

    stage(0, 0);
    __syncthreads();
    for (int kt = 0; kt < NKT; ++kt) {
        const int cur = kt & 1;
        if (kt + 1 < NKT) stage(kt + 1, cur ^ 1);   // prefetch into alt buffer
        const v16h a = load_frag(&As[cur][(16*w + ln)*LDA], g);
        #pragma unroll
        for (int f = 0; f < NF; ++f) {
            const v16h b = load_frag(&Bs[cur][(16*f + ln)*LDA], g);
            acc[f] = wmma32f16(a, b, acc[f]);
        }
        __syncthreads();   // staging done for alt buffer; reads of cur done
    }

    // ---- epilogue: bias + store ----
    float*    Cf = (float*)Cp;
    _Float16* Ch = (_Float16*)Cp;
    #pragma unroll
    for (int f = 0; f < NF; ++f) {
        const int nn = n0 + 16*f + ln;
        const float bv = bias[nn];
        #pragma unroll
        for (int vv = 0; vv < 8; ++vv) {
            const int m = m0 + 16*w + 8*g + vv;       // C layout: m = v + 8*g
            const float val = acc[f][vv] + bv;
            if (SPLIT) {
                const int b = m >> 11;                 // / TSEQ
                const int t = m & (TSEQ - 1);
                const int h = nn >> 6;                 // / HD
                const int d = nn & (HD - 1);
                Ch[((((size_t)b*NH + h)*TSEQ + t) << 6) + d] = (_Float16)val;
            } else {
                Cf[(size_t)m * GN + nn] = val;
            }
        }
    }
}

// ---------------- flash attention (causal, online softmax) ----------------
// grid: (T/64, H, B); block: 128 (4 waves, 16 q-rows each).
#define LVT 72   // padded LDS stride (halves); 144B rows keep 16B alignment

__global__ __launch_bounds__(128)
void wmma_flash_attn(const _Float16* __restrict__ Q, const _Float16* __restrict__ Kq,
                     const _Float16* __restrict__ V, _Float16* __restrict__ Oh)
{
    __shared__ _Float16 ldsVt[HD * LVT];       // V^T tile: [d][n]
    __shared__ _Float16 ldsP[4 * 16 * LVT];    // per-wave P tiles: [m][n]

    const int tid  = threadIdx.x;
    const int w    = tid >> 5;
    const int lane = tid & 31;
    const int g    = lane >> 4;
    const int ln   = lane & 15;
    const int q0   = blockIdx.x * 64;
    const size_t bh = ((size_t)blockIdx.z * NH + blockIdx.y) * (size_t)TSEQ * HD;

    // Q fragments stay resident (A-layout, per-lane contiguous in d)
    const int tq = q0 + 16*w + ln;
    const _Float16* qrow = Q + bh + (size_t)tq * HD;
    const v16h qf0 = load_frag(qrow,       g);
    const v16h qf1 = load_frag(qrow + 32,  g);

    const v8f zf = {0.f,0.f,0.f,0.f,0.f,0.f,0.f,0.f};
    float m_i[8], l_i[8];
    v8f o[4]; o[0]=zf; o[1]=zf; o[2]=zf; o[3]=zf;
    #pragma unroll
    for (int vv = 0; vv < 8; ++vv) { m_i[vv] = -1e30f; l_i[vv] = 0.f; }

    _Float16* Pw = ldsP + w * 16 * LVT;

    for (int j = 0; j <= (int)blockIdx.x; ++j) {   // causal: only tiles n0 <= q0+63
        const int n0 = j * 64;
        __syncthreads();                            // ldsVt reuse guard
        // ---- stage V^T tile ----
        for (int e = tid; e < 64*8; e += 128) {
            const int n  = e >> 3;
            const int db = (e & 7) * 8;
            const v8h x = *(const v8h*)(V + bh + (size_t)(n0 + n)*HD + db);
            #pragma unroll
            for (int t2 = 0; t2 < 8; ++t2) ldsVt[(db + t2)*LVT + n] = x[t2];
        }
        __syncthreads();

        // ---- S = Q @ K^T  (B-frags straight from global K, contiguous in d) ----
        v8f s[4]; s[0]=zf; s[1]=zf; s[2]=zf; s[3]=zf;
        #pragma unroll
        for (int ks = 0; ks < 2; ++ks) {
            const v16h a = ks ? qf1 : qf0;
            #pragma unroll
            for (int f = 0; f < 4; ++f) {
                const _Float16* kp = Kq + bh + (size_t)(n0 + 16*f + ln)*HD + 32*ks;
                const v16h b = load_frag(kp, g);
                s[f] = wmma32f16(a, b, s[f]);
            }
        }
        // ---- scale + causal mask ----
        #pragma unroll
        for (int f = 0; f < 4; ++f) {
            const int ng = n0 + 16*f + ln;
            #pragma unroll
            for (int vv = 0; vv < 8; ++vv) {
                const int mg = q0 + 16*w + 8*g + vv;
                const float sv = s[f][vv] * SCALE_F;
                s[f][vv] = (ng > mg) ? -1e30f : sv;
            }
        }
        // ---- online softmax (rows live in 16-lane halves; xor-shuffle reduce) ----
        float mnew[8], alpha[8], rs[8];
        #pragma unroll
        for (int vv = 0; vv < 8; ++vv) {
            float r = fmaxf(fmaxf(s[0][vv], s[1][vv]), fmaxf(s[2][vv], s[3][vv]));
            r = fmaxf(r, __shfl_xor(r, 1, 32));
            r = fmaxf(r, __shfl_xor(r, 2, 32));
            r = fmaxf(r, __shfl_xor(r, 4, 32));
            r = fmaxf(r, __shfl_xor(r, 8, 32));
            const float mn = fmaxf(m_i[vv], r);
            mnew[vv] = mn; alpha[vv] = __expf(m_i[vv] - mn); m_i[vv] = mn; rs[vv] = 0.f;
        }
        #pragma unroll
        for (int f = 0; f < 4; ++f) {
            const int nl = 16*f + ln;
            #pragma unroll
            for (int vv = 0; vv < 8; ++vv) {
                const float p = __expf(s[f][vv] - mnew[vv]);
                rs[vv] += p;
                Pw[(8*g + vv)*LVT + nl] = (_Float16)p;   // C-layout -> [m][n] in LDS
            }
        }
        #pragma unroll
        for (int vv = 0; vv < 8; ++vv) {
            float r = rs[vv];
            r += __shfl_xor(r, 1, 32);
            r += __shfl_xor(r, 2, 32);
            r += __shfl_xor(r, 4, 32);
            r += __shfl_xor(r, 8, 32);
            l_i[vv] = l_i[vv]*alpha[vv] + r;
        }
        #pragma unroll
        for (int fd = 0; fd < 4; ++fd)
            #pragma unroll
            for (int vv = 0; vv < 8; ++vv)
                o[fd][vv] *= alpha[vv];
        // ---- O += P @ V  (A-frags from per-wave LDS P, B-frags from V^T) ----
        #pragma unroll
        for (int ks = 0; ks < 2; ++ks) {
            const v16h a = load_frag(Pw + ln*LVT + 32*ks, g);
            #pragma unroll
            for (int fd = 0; fd < 4; ++fd) {
                const v16h b = load_frag(ldsVt + (16*fd + ln)*LVT + 32*ks, g);
                o[fd] = wmma32f16(a, b, o[fd]);
            }
        }
    }

    // ---- normalize and write (B,T,D) f16 for final projection ----
    #pragma unroll
    for (int vv = 0; vv < 8; ++vv) {
        const float inv = 1.0f / l_i[vv];
        const int t2 = q0 + 16*w + 8*g + vv;
        const size_t row = ((size_t)blockIdx.z * TSEQ + t2) * (size_t)DMODEL
                         + (size_t)blockIdx.y * HD;
        #pragma unroll
        for (int fd = 0; fd < 4; ++fd)
            Oh[row + 16*fd + ln] = (_Float16)(o[fd][vv] * inv);
    }
}

// ---------------- launch ----------------
extern "C" void kernel_launch(void* const* d_in, const int* in_sizes, int n_in,
                              void* d_out, int out_size, void* d_ws, size_t ws_size,
                              hipStream_t stream) {
    (void)in_sizes; (void)n_in; (void)out_size; (void)ws_size;
    const float* Xq = (const float*)d_in[0];
    const float* Xk = (const float*)d_in[1];
    const float* Xv = (const float*)d_in[2];
    const float* Wq = (const float*)d_in[3];
    const float* bq = (const float*)d_in[4];
    const float* Wk = (const float*)d_in[5];
    const float* bk = (const float*)d_in[6];
    const float* Wv = (const float*)d_in[7];
    const float* bv = (const float*)d_in[8];
    const float* Wo = (const float*)d_in[9];
    const float* bo = (const float*)d_in[10];
    float* out = (float*)d_out;

    // workspace: q,k,v in (B,H,T,Hd) f16, attn in (B,T,D) f16
    const size_t MAT = (size_t)GM * GK;   // 8M halves each
    _Float16* qh = (_Float16*)d_ws;
    _Float16* kh = qh + MAT;
    _Float16* vh = qh + 2*MAT;
    _Float16* ah = qh + 3*MAT;

    const dim3 gg(GN / BN, GM / BM);      // (8, 64)
    wmma_gemm_bias<false, true ><<<gg, 256, 0, stream>>>(Xq, Wq, bq, qh);
    wmma_gemm_bias<false, true ><<<gg, 256, 0, stream>>>(Xk, Wk, bk, kh);
    wmma_gemm_bias<false, true ><<<gg, 256, 0, stream>>>(Xv, Wv, bv, vh);

    wmma_flash_attn<<<dim3(TSEQ/64, NH, BB), 128, 0, stream>>>(qh, kh, vh, ah);

    wmma_gemm_bias<true, false><<<gg, 256, 0, stream>>>(ah, Wo, bo, out);
}
